// SSMALayer_79388175499425
// MI455X (gfx1250) — matrix operand order repaired and checked by reading, not compiled, not run on hardware
//
#include <hip/hip_runtime.h>
#include <stdint.h>

// ---------------------------------------------------------------------------
// CDNA5 (gfx1250) implementation of the SSMA layer.
//   - FFN + input projection: bf16 WMMA tiled GEMMs (v_wmma_f32_16x16x32_bf16)
//     with double-buffered Tensor-Data-Mover staging (tensor_load_to_lds,
//     counted s_wait_tensorcnt for DMA/compute overlap)
//   - Recurrent scan: single persistent WGP, w_state resident in 320KB LDS
// ---------------------------------------------------------------------------

typedef __attribute__((ext_vector_type(16))) __bf16 v16bf;
typedef __attribute__((ext_vector_type(8)))  float  v8f;
typedef __attribute__((ext_vector_type(4)))  unsigned int v4u;
typedef __attribute__((ext_vector_type(4)))  int    v4i;
typedef __attribute__((ext_vector_type(8)))  int    v8i;

#define BM 128
#define BN 128
#define BK 32
#define LDK 40   // padded LDS row stride in halfs (80B, 16B-aligned, conflict-free)
#define ATILE (BM * LDK)   // halfs per A stage buffer
#define BTILE (BN * LDK)   // halfs per B stage buffer

// TDM availability (device pass only; host pass falls back silently)
#if defined(__has_builtin)
#  if __has_builtin(__builtin_amdgcn_tensor_load_to_lds) && \
      __has_builtin(__builtin_amdgcn_s_wait_tensorcnt)
#    define USE_TDM 1
#  endif
#endif
#ifndef USE_TDM
#  define USE_TDM 0
#endif

__device__ __forceinline__ unsigned short f32_to_bf16_rne(float f) {
  union { float f; uint32_t u; } c; c.f = f;
  uint32_t u = c.u;
  u += 0x7FFFu + ((u >> 16) & 1u);
  return (unsigned short)(u >> 16);
}

union Frag { uint4 q[2]; v16bf v; };

#if USE_TDM
// Issue one TDM 2D-tile load: rows x BK bf16 tile from row-major global
// (row stride = strideElems) into LDS at lds_byte_off, with hardware row
// padding of 4 DWORDs after every 16 DWORDs (32 halfs -> LDK=40 halfs).
__device__ __forceinline__ void tdm_load_tile_bf16(const unsigned short* gptr,
                                                   uint32_t lds_byte_off,
                                                   uint32_t rows, uint32_t cols,
                                                   uint32_t tensor_rows,
                                                   uint32_t tensor_cols,
                                                   uint64_t strideElems) {
  uint64_t ga = (uint64_t)(uintptr_t)gptr;
  v4u g0;
  g0.x = 1u;                                      // count=1, is_restore=0
  g0.y = lds_byte_off;                            // [63:32] lds_addr
  g0.z = (uint32_t)(ga & 0xFFFFFFFFu);            // [95:64] global_addr lo
  g0.w = (uint32_t)((ga >> 32) & 0x01FFFFFFu)     // [120:96] global_addr hi
         | (2u << 30);                            // [127:126] type=2 (image)
  v8i g1;
  // w0: wg_mask=0 | data_size=1 (2B) | pad_enable | pad_interval=3 (16 DW)
  //     | pad_amount=3 (4 DW)
  g1[0] = (int)((1u << 16) | (1u << 20) | (3u << 22) | (3u << 25));
  g1[1] = (int)((tensor_cols & 0xFFFFu) << 16);                    // td0 lo16
  g1[2] = (int)(((tensor_cols >> 16) & 0xFFFFu) |
                ((tensor_rows & 0xFFFFu) << 16));                  // td0 hi/td1 lo
  g1[3] = (int)(((tensor_rows >> 16) & 0xFFFFu) | (cols << 16));   // td1 hi/tile0
  g1[4] = (int)(rows & 0xFFFFu);                                   // tile1, tile2=0
  g1[5] = (int)(uint32_t)(strideElems & 0xFFFFFFFFu);              // stride0 lo32
  g1[6] = (int)(uint32_t)((strideElems >> 32) & 0xFFFFu);          // stride0 hi16
  g1[7] = 0;
  v4i z4 = {0, 0, 0, 0};
#if __clang_major__ >= 23
  v8i z8 = {0, 0, 0, 0, 0, 0, 0, 0};
  __builtin_amdgcn_tensor_load_to_lds(g0, g1, z4, z4, z8, 0);
#else
  __builtin_amdgcn_tensor_load_to_lds(g0, g1, z4, z4, 0);
#endif
}
#endif

// ------------------------------- convert -----------------------------------
__global__ __launch_bounds__(256)
void convert_f32_to_bf16(const float* __restrict__ in,
                         unsigned short* __restrict__ out, int n) {
  int i = blockIdx.x * blockDim.x + threadIdx.x;
  if (i < n) out[i] = f32_to_bf16_rne(in[i]);
}

// ------------------------------ NT GEMM ------------------------------------
// C[m,n] = sum_k A[m,k] * Bm[n,k]  (both row-major along K, bf16)
// mode 0: Cf  = acc + bias[n]                       (f32 out)
// mode 1: Cbf = bf16(relu(acc + bias[n]))           (bf16 out)
// mode 2: Cf  = acc + bias[n] + resid[m,n]          (f32 out)
__global__ __launch_bounds__(256)
void gemm_nt_bf16_wmma(const unsigned short* __restrict__ A,
                       const unsigned short* __restrict__ Bm,
                       const float* __restrict__ bias,
                       const float* __restrict__ resid,
                       float* __restrict__ Cf,
                       unsigned short* __restrict__ Cbf,
                       int Mdim, int Ndim, int Kdim, int mode)
{
  __shared__ unsigned short shA[2 * ATILE];   // double-buffered A stages
  __shared__ unsigned short shB[2 * BTILE];   // double-buffered B stages

  const int tid  = threadIdx.x;
  const int lane = tid & 31;
  const int hl   = lane >> 4;     // half-wave select
  const int r    = lane & 15;
  const int wv   = tid >> 5;      // 8 waves
  const int wvM  = wv >> 1;       // 0..3 -> 32-row strips
  const int wvN  = wv & 1;        // 0..1 -> 64-col strips

  const int blockM = blockIdx.y * BM;
  const int blockN = blockIdx.x * BN;

  const v8f vzero = {0.f,0.f,0.f,0.f,0.f,0.f,0.f,0.f};
  v8f acc[2][4];
#pragma unroll
  for (int mi = 0; mi < 2; ++mi)
#pragma unroll
    for (int ni = 0; ni < 4; ++ni) acc[mi][ni] = vzero;

  const int nstage = Kdim / BK;

#if USE_TDM
  const uint32_t shA_off = (uint32_t)(uintptr_t)(void*)shA;
  const uint32_t shB_off = (uint32_t)(uintptr_t)(void*)shB;
  const unsigned short* Atile = A  + (size_t)blockM * Kdim;
  const unsigned short* Btile = Bm + (size_t)blockN * Kdim;

  // prologue: stage 0 DMA into buffer 0 (wave 0 drives the tensor engine)
  if (tid < 32) {
    tdm_load_tile_bf16(Atile, shA_off, BM, BK, BM, Kdim, Kdim);
    tdm_load_tile_bf16(Btile, shB_off, BN, BK, BN, Kdim, Kdim);
  }
#else
  // cooperative tile loads: 2 threads per row, 16B each
  const int lrow = tid >> 1;
  const int lsel = tid & 1;
  const unsigned short* Aptr = A  + (size_t)(blockM + lrow) * Kdim + lsel * 16;
  const unsigned short* Bptr = Bm + (size_t)(blockN + lrow) * Kdim + lsel * 16;
  uint4* sa = (uint4*)&shA[lrow * LDK + lsel * 16];
  uint4* sb = (uint4*)&shB[lrow * LDK + lsel * 16];
#endif

  for (int i = 0; i < nstage; ++i) {
#if USE_TDM
    const int cur = i & 1;
    __syncthreads();                    // buf[cur^1] consumers (stage i-1) done
    if (tid < 32) {
      if (i + 1 < nstage) {             // overlap: next stage DMA in flight
        tdm_load_tile_bf16(Atile + (size_t)(i + 1) * BK,
                           shA_off + (uint32_t)(cur ^ 1) * (ATILE * 2),
                           BM, BK, BM, Kdim, Kdim);
        tdm_load_tile_bf16(Btile + (size_t)(i + 1) * BK,
                           shB_off + (uint32_t)(cur ^ 1) * (BTILE * 2),
                           BN, BK, BN, Kdim, Kdim);
        __builtin_amdgcn_s_wait_tensorcnt(2);  // stage i complete, i+1 in flight
      } else {
        __builtin_amdgcn_s_wait_tensorcnt(0);  // drain last stage
      }
    }
    __syncthreads();
    const unsigned short* sA = &shA[cur * ATILE];
    const unsigned short* sB = &shB[cur * BTILE];
#else
    const int k0 = i * BK;
    const uint4* ag = (const uint4*)(Aptr + k0);
    const uint4* bg = (const uint4*)(Bptr + k0);
    uint4 a0 = ag[0];
    uint4 b0 = bg[0];
    if (k0 + BK < Kdim) {               // pull next K-slab toward GL2/WGP$
      __builtin_prefetch(Aptr + k0 + BK, 0, 1);
      __builtin_prefetch(Bptr + k0 + BK, 0, 1);
    }
    __syncthreads();                    // previous stage fully consumed
    sa[0] = a0;
    sb[0] = b0;
    __syncthreads();
    const unsigned short* sA = shA;
    const unsigned short* sB = shB;
#endif

    {
      Frag afr[2], bfr[4];
      // A fragment: lane(hl,r) holds row r, K = 16*hl..16*hl+15 (contiguous)
#pragma unroll
      for (int mi = 0; mi < 2; ++mi) {
        int row = wvM * 32 + mi * 16 + r;
        afr[mi].q[0] = *(const uint4*)&sA[row * LDK + 16 * hl];
        afr[mi].q[1] = *(const uint4*)&sA[row * LDK + 16 * hl + 8];
      }
      // B fragment: lane(hl,r) holds col r, K = 16*hl..16*hl+15 (contiguous)
#pragma unroll
      for (int ni = 0; ni < 4; ++ni) {
        int col = wvN * 64 + ni * 16 + r;
        const uint4* p = (const uint4*)&sB[col * LDK + 16 * hl];
        bfr[ni].q[0] = p[0];
        bfr[ni].q[1] = p[1];
      }
#pragma unroll
      for (int mi = 0; mi < 2; ++mi)
#pragma unroll
        for (int ni = 0; ni < 4; ++ni)
          acc[mi][ni] = __builtin_amdgcn_wmma_f32_16x16x32_bf16(
              false, afr[mi].v, false, bfr[ni].v,
              (short)0, acc[mi][ni], false, false);
    }
  }

  // epilogue: C tile layout -> lane(hl,r): col = r, rows = 8*hl + v
  // mode branch hoisted outside the fully-unrolled loops (constant vector
  // indices -> no scratch spills)
  const int colBase = blockN + wvN * 64 + r;
  const int rowBase = blockM + wvM * 32 + 8 * hl;
  if (mode == 1) {
#pragma unroll
    for (int mi = 0; mi < 2; ++mi)
#pragma unroll
      for (int ni = 0; ni < 4; ++ni) {
        int col = colBase + ni * 16;
        float bv = bias[col];
#pragma unroll
        for (int v = 0; v < 8; ++v) {
          float val = acc[mi][ni][v] + bv;
          size_t idx = (size_t)(rowBase + mi * 16 + v) * Ndim + col;
          Cbf[idx] = f32_to_bf16_rne(val > 0.f ? val : 0.f);
        }
      }
  } else if (mode == 2) {
#pragma unroll
    for (int mi = 0; mi < 2; ++mi)
#pragma unroll
      for (int ni = 0; ni < 4; ++ni) {
        int col = colBase + ni * 16;
        float bv = bias[col];
#pragma unroll
        for (int v = 0; v < 8; ++v) {
          size_t idx = (size_t)(rowBase + mi * 16 + v) * Ndim + col;
          Cf[idx] = acc[mi][ni][v] + bv + resid[idx];
        }
      }
  } else {
#pragma unroll
    for (int mi = 0; mi < 2; ++mi)
#pragma unroll
      for (int ni = 0; ni < 4; ++ni) {
        int col = colBase + ni * 16;
        float bv = bias[col];
#pragma unroll
        for (int v = 0; v < 8; ++v) {
          size_t idx = (size_t)(rowBase + mi * 16 + v) * Ndim + col;
          Cf[idx] = acc[mi][ni][v] + bv;
        }
      }
  }
}

// ------------------------------- scan --------------------------------------
// One persistent workgroup; w_state (256KB, transposed) lives in LDS.
__global__ __launch_bounds__(256)
void ssma_scan(const float* __restrict__ xproj,   // [B, S, M] f32
               const float* __restrict__ state,   // [B, M]
               const float* __restrict__ w_state, // [M, M]
               const float* __restrict__ gamma_p,
               const int*   __restrict__ topk_p,
               float* __restrict__ mem_out,       // [B, M]
               int S)
{
  extern __shared__ float smem[];
  float* wT  = smem;                 // 256*256, transposed: wT[k*256+m]
  float* st  = wT  + 256 * 256;      // 8*256
  float* mem = st  + 8 * 256;        // 8*256
  float* su  = mem + 8 * 256;        // 8*256

  const int tid  = threadIdx.x;
  const int lane = tid & 31;
  const int wv   = tid >> 5;         // wave wv owns batch row b = wv
  const float gamma = *gamma_p;
  const int   K     = *topk_p;

  for (int i = tid; i < 256 * 256; i += 256) {
    int m = i >> 8, k = i & 255;
    wT[k * 256 + m] = w_state[i];
  }
  for (int i = tid; i < 8 * 256; i += 256) { st[i] = state[i]; mem[i] = state[i]; }
  __syncthreads();

  const int m = tid;                 // this thread owns column m
  for (int t = 0; t < S; ++t) {
    float accv[8];
#pragma unroll
    for (int b = 0; b < 8; ++b) accv[b] = 0.f;
    for (int k = 0; k < 256; ++k) {
      float w = wT[k * 256 + m];                 // conflict-free
#pragma unroll
      for (int b = 0; b < 8; ++b) accv[b] += st[b * 256 + k] * w;  // broadcast
    }
#pragma unroll
    for (int b = 0; b < 8; ++b) {
      float v = accv[b] + xproj[((size_t)b * S + t) * 256 + m];
      su[b * 256 + m] = v > 0.f ? v : 0.f;
    }
    __syncthreads();

    // per-row K-th largest via bitwise binary search (values >= 0)
    {
      uint32_t bits[8];
#pragma unroll
      for (int j = 0; j < 8; ++j) {
        union { float f; uint32_t u; } c; c.f = su[wv * 256 + lane + 32 * j];
        bits[j] = c.u;
      }
      uint32_t thr = 0;
      for (int bit = 30; bit >= 0; --bit) {
        uint32_t cand = thr | (1u << bit);
        int cnt = 0;
#pragma unroll
        for (int j = 0; j < 8; ++j) cnt += (bits[j] >= cand) ? 1 : 0;
        for (int off = 16; off > 0; off >>= 1) cnt += __shfl_xor(cnt, off, 32);
        if (cnt >= K) thr = cand;
      }
#pragma unroll
      for (int j = 0; j < 8; ++j) {
        int mm = wv * 256 + lane + 32 * j;
        union { uint32_t u; float f; } c; c.u = bits[j];
        float g = (bits[j] >= thr) ? c.f : 0.f;
        st[mm]  = g;
        mem[mm] = gamma * mem[mm] + (1.f - gamma) * g;
      }
    }
    __syncthreads();
  }

  for (int i = tid; i < 8 * 256; i += 256) mem_out[i] = mem[i];
}

// ------------------------------ launcher -----------------------------------
extern "C" void kernel_launch(void* const* d_in, const int* in_sizes, int n_in,
                              void* d_out, int out_size, void* d_ws, size_t ws_size,
                              hipStream_t stream) {
  (void)in_sizes; (void)n_in; (void)out_size; (void)ws_size;
  const int B = 8, S = 2048, D = 1024, M = 256, F = 4096;
  const int BS = B * S;

  const float* x       = (const float*)d_in[0];
  const float* state   = (const float*)d_in[1];
  const float* w_in_w  = (const float*)d_in[2];
  const float* w_in_b  = (const float*)d_in[3];
  const float* w_state = (const float*)d_in[4];
  const float* ffn_w1  = (const float*)d_in[5];
  const float* ffn_b1  = (const float*)d_in[6];
  const float* ffn_w2  = (const float*)d_in[7];
  const float* ffn_b2  = (const float*)d_in[8];
  const float* gamma_p = (const float*)d_in[9];
  const int*   topk_p  = (const int*)d_in[10];

  float* out_f   = (float*)d_out;                    // [B,S,D] outputs
  float* out_mem = out_f + (size_t)BS * D;           // [B,M] memory

  // workspace layout
  char* ws = (char*)d_ws;
  unsigned short* xbf   = (unsigned short*)ws;                 ws += (size_t)BS * D * 2;
  unsigned short* winbf = (unsigned short*)ws;                 ws += (size_t)M * D * 2;
  unsigned short* w1bf  = (unsigned short*)ws;                 ws += (size_t)F * D * 2;
  unsigned short* w2bf  = (unsigned short*)ws;                 ws += (size_t)D * F * 2;
  float*          xproj = (float*)ws;                          ws += (size_t)BS * M * 4;
  unsigned short* hbf   = (unsigned short*)ws;                 /* BS*F bf16 */

  // 1) fp32 -> bf16 conversions
  {
    int n;
    n = BS * D; convert_f32_to_bf16<<<(n + 255) / 256, 256, 0, stream>>>(x, xbf, n);
    n = M * D;  convert_f32_to_bf16<<<(n + 255) / 256, 256, 0, stream>>>(w_in_w, winbf, n);
    n = F * D;  convert_f32_to_bf16<<<(n + 255) / 256, 256, 0, stream>>>(ffn_w1, w1bf, n);
    n = D * F;  convert_f32_to_bf16<<<(n + 255) / 256, 256, 0, stream>>>(ffn_w2, w2bf, n);
  }

  // 2) x_proj = x @ w_in_w^T + b   (f32 out)
  gemm_nt_bf16_wmma<<<dim3(M / BN, BS / BM), 256, 0, stream>>>(
      xbf, winbf, w_in_b, nullptr, xproj, nullptr, BS, M, D, 0);

  // 3) scan -> memory (depends only on x_proj; serialized by stream order)
  {
    size_t smem = (size_t)(256 * 256 + 3 * 8 * 256) * sizeof(float); // 280KB < 320KB WGP LDS
    (void)hipFuncSetAttribute((const void*)ssma_scan,
                              hipFuncAttributeMaxDynamicSharedMemorySize, (int)smem);
    ssma_scan<<<1, 256, smem, stream>>>(xproj, state, w_state, gamma_p, topk_p, out_mem, S);
  }

  // 4) h = relu(x @ w1^T + b1)   (bf16 out)
  gemm_nt_bf16_wmma<<<dim3(F / BN, BS / BM), 256, 0, stream>>>(
      xbf, w1bf, ffn_b1, nullptr, nullptr, hbf, BS, F, D, 1);

  // 5) out = x + h @ w2^T + b2   (f32 out)
  gemm_nt_bf16_wmma<<<dim3(D / BN, BS / BM), 256, 0, stream>>>(
      hbf, w2bf, ffn_b2, x, out_f, nullptr, BS, D, F, 2);
}
